// MobiusLinear_13091060318399
// MI455X (gfx1250) — compile-verified, matrix operand order checked by reading, not run on hardware
//
#include <hip/hip_runtime.h>
#include <math.h>

typedef __attribute__((ext_vector_type(16))) _Float16 v16h;
typedef __attribute__((ext_vector_type(8)))  _Float16 v8h;
typedef __attribute__((ext_vector_type(8)))  float    v8f;
typedef __attribute__((ext_vector_type(4)))  float    v4f;

#define DIN         128
#define DOUT        128
#define ROWS_PER_WG 128
#define THREADS     256
#define W_STRIDE    136   // halfs per W row in LDS (272B: 16B-aligned, bank-staggered)
#define MX_STRIDE   132   // floats per Mx row in LDS (528B: 16B-aligned)

__global__ __launch_bounds__(THREADS)
void mobius_linear_kernel(const float* __restrict__ x,
                          const float* __restrict__ W,
                          const float* __restrict__ bias,
                          float* __restrict__ out,
                          int Btotal)
{
    __shared__ __align__(16) _Float16 sW[DOUT * W_STRIDE];          // 34816 B
    __shared__ __align__(16) float    sMx[ROWS_PER_WG * MX_STRIDE]; // 67584 B
    __shared__ __align__(16) float    sBias[DOUT];
    __shared__ float sXn2[ROWS_PER_WG];
    __shared__ float sAlpha[ROWS_PER_WG];
    __shared__ float sBeta[ROWS_PER_WG];
    __shared__ float sP[3][THREADS];                                 // 3072 B partials

    const int tid  = threadIdx.x;
    const int lane = tid & 31;
    const int wave = tid >> 5;
    const int wgRow0 = blockIdx.x * ROWS_PER_WG;
    if (wgRow0 >= Btotal) return;

    // ---- Prefetch this wave's x rows into cache while W is staged ----
    {
        int prow = wgRow0 + wave * 16 + (lane & 15);
        if (prow >= Btotal) prow = Btotal - 1;
        __builtin_prefetch(&x[(size_t)prow * DIN], 0, 3);
        __builtin_prefetch(&x[(size_t)prow * DIN + 64], 0, 3);
    }

    // ---- Stage W (fp32 -> f16) into padded LDS; stage bias ----
    {
        const v4f* W4 = (const v4f*)W;
        #pragma unroll
        for (int i = tid; i < DOUT * DIN / 4; i += THREADS) {
            const int flat = i * 4;
            const int n = flat >> 7;
            const int k = flat & 127;
            v4f w = W4[i];
            _Float16* dst = &sW[n * W_STRIDE + k];
            dst[0] = (_Float16)w.x; dst[1] = (_Float16)w.y;
            dst[2] = (_Float16)w.z; dst[3] = (_Float16)w.w;
        }
        if (tid < DOUT) sBias[tid] = bias[tid];
    }

    // ---- Load A fragments (16-bit 16x32 MxK layout) + fp32 row sum-of-squares ----
    const int m  = lane & 15;                 // row within 16-row tile
    const int kb = (lane & 16) ? 8 : 0;       // K sub-base per half-wave
    int rowIdx = wgRow0 + wave * 16 + m;
    if (rowIdx >= Btotal) rowIdx = Btotal - 1;   // clamp (EXEC stays all-ones for WMMA)
    const size_t rowBase = (size_t)rowIdx * DIN;

    v16h a[4];
    float sumsq = 0.0f;
    #pragma unroll
    for (int kc = 0; kc < 4; ++kc) {
        const int k0 = kc * 32 + kb;
        v4f f0 = *(const v4f*)&x[rowBase + k0];        // K = k0   .. k0+3
        v4f f1 = *(const v4f*)&x[rowBase + k0 + 4];    // K = k0+4 .. k0+7
        v4f f2 = *(const v4f*)&x[rowBase + k0 + 16];   // K = k0+16.. k0+19
        v4f f3 = *(const v4f*)&x[rowBase + k0 + 20];   // K = k0+20.. k0+23
        #pragma unroll
        for (int j = 0; j < 4; ++j) {
            a[kc][j]      = (_Float16)f0[j];
            a[kc][4 + j]  = (_Float16)f1[j];
            a[kc][8 + j]  = (_Float16)f2[j];
            a[kc][12 + j] = (_Float16)f3[j];
            sumsq += f0[j]*f0[j] + f1[j]*f1[j] + f2[j]*f2[j] + f3[j]*f3[j];
        }
    }
    // combine the two half-wave partials of the same row (wave32 xor-16 shuffle)
    sumsq += __shfl_xor(sumsq, 16, 32);
    if (lane < 16) sXn2[wave * 16 + lane] = sumsq;

    __syncthreads();   // sW / sBias ready

    // ---- GEMM tile: Mx[16 x 128]; per N-tile, preload ALL 4 B fragments into
    //      distinct registers so ds_load latency overlaps the WMMA chain ----
    const int rl0 = wave * 16 + ((lane & 16) ? 8 : 0); // local row base for D fragment
    #pragma unroll
    for (int nt = 0; nt < 8; ++nt) {
        const _Float16* wrow = &sW[(nt * 16 + m) * W_STRIDE + kb];
        v16h b[4];
        #pragma unroll
        for (int kc = 0; kc < 4; ++kc) {
            v8h lo = *(const v8h*)(wrow + kc * 32);        // K = k0   .. k0+7
            v8h hi = *(const v8h*)(wrow + kc * 32 + 16);   // K = k0+16.. k0+23
            b[kc] = __builtin_shufflevector(lo, hi,
                        0,1,2,3,4,5,6,7,8,9,10,11,12,13,14,15);
        }
        v8f acc = {0.f,0.f,0.f,0.f,0.f,0.f,0.f,0.f};
        #pragma unroll
        for (int kc = 0; kc < 4; ++kc)
            acc = __builtin_amdgcn_wmma_f32_16x16x32_f16(
                      false, a[kc], false, b[kc], (short)0, acc, false, false);

        const int col = nt * 16 + m;
        #pragma unroll
        for (int v = 0; v < 8; ++v)
            sMx[(rl0 + v) * MX_STRIDE + col] = acc[v];   // consecutive lanes -> consecutive banks
    }

    __syncthreads();   // Mx tile + xnorms ready

    // ---- Row reductions with all 256 threads: 2 threads/row x 64 cols, float4 ----
    {
        const int r = tid & 127;
        const int h = tid >> 7;
        const v4f* mrow = (const v4f*)&sMx[r * MX_STRIDE + h * 64];
        const v4f* brow = (const v4f*)&sBias[h * 64];
        float s1 = 0.f, s2 = 0.f, s3 = 0.f;   // ||Mx||^2, Mx.b, ||b||^2 (partials)
        #pragma unroll
        for (int q = 0; q < 16; ++q) {
            v4f mv = mrow[q];
            v4f bv = brow[q];
            #pragma unroll
            for (int e = 0; e < 4; ++e) {
                s1 = fmaf(mv[e], mv[e], s1);
                s2 = fmaf(mv[e], bv[e], s2);
                s3 = fmaf(bv[e], bv[e], s3);
            }
        }
        sP[0][tid] = s1; sP[1][tid] = s2; sP[2][tid] = s3;
    }
    __syncthreads();

    // ---- Per-row scalar epilogue (threads 0..127 own one row each) ----
    if (tid < ROWS_PER_WG) {
        const float s1 = sP[0][tid] + sP[0][tid + 128];
        const float s2 = sP[1][tid] + sP[1][tid + 128];
        const float y2 = sP[2][tid] + sP[2][tid + 128];   // ||b||^2
        float xn  = fmaxf(sqrtf(sXn2[tid]), 1e-15f);
        float Mxn = fmaxf(sqrtf(s1), 1e-15f);
        // artanh(clip(xn, .., 1-1e-7))
        float u  = fminf(xn, 1.0f - 1e-7f);
        float at = 0.5f * (log1pf(u) - log1pf(-u));
        float sc = tanhf((Mxn / xn) * at) / Mxn;   // mobius_matvec scale (c = 1)
        if (Mxn <= 1e-10f) sc = 0.0f;
        // mobius_add(y = sc*Mx, bias): closed-form coefficients
        const float x2 = sc * sc * s1;             // ||y||^2
        const float xy = sc * s2;                  // y.b
        const float den = fmaxf(1.0f + 2.0f*xy + x2*y2, 1e-15f);
        const float cA  = (1.0f + 2.0f*xy + y2) / den;
        const float cB  = (1.0f - x2) / den;
        // ||res||^2 in closed form -> projx without another pass
        const float rn2 = cA*cA*x2 + 2.0f*cA*cB*xy + cB*cB*y2;
        const float rn  = fmaxf(sqrtf(rn2), 1e-15f);
        const float maxn = 1.0f - 1e-5f;
        const float f = (rn > maxn) ? (maxn / rn) : 1.0f;
        sAlpha[tid] = f * cA * sc;   // multiplies Mx
        sBeta[tid]  = f * cB;        // multiplies bias
    }

    __syncthreads();

    // ---- Coalesced vectorized write: out = alpha*Mx + beta*bias ----
    {
        const int nvec = ROWS_PER_WG * DOUT / 4;   // 4096 float4s
        v4f* out4 = (v4f*)(out + (size_t)wgRow0 * DOUT);
        const int rowsLeft = Btotal - wgRow0;
        #pragma unroll
        for (int i = tid; i < nvec; i += THREADS) {
            const int flat = i * 4;
            const int r = flat >> 7;
            const int j = flat & 127;
            if (r >= rowsLeft) break;
            const float al = sAlpha[r];
            const float be = sBeta[r];
            v4f mv = *(const v4f*)&sMx[r * MX_STRIDE + j];
            v4f bv = *(const v4f*)&sBias[j];
            out4[i] = al * mv + be * bv;
        }
    }
}

extern "C" void kernel_launch(void* const* d_in, const int* in_sizes, int n_in,
                              void* d_out, int out_size, void* d_ws, size_t ws_size,
                              hipStream_t stream) {
    const float* x    = (const float*)d_in[0];
    const float* W    = (const float*)d_in[1];
    const float* bias = (const float*)d_in[2];
    float* out        = (float*)d_out;
    const int Btotal  = in_sizes[0] / DIN;
    const int grid    = (Btotal + ROWS_PER_WG - 1) / ROWS_PER_WG;
    hipLaunchKernelGGL(mobius_linear_kernel, dim3(grid), dim3(THREADS), 0, stream,
                       x, W, bias, out, Btotal);
}
